// MME_80169859547285
// MI455X (gfx1250) — matrix operand order repaired and checked by reading, not compiled, not run on hardware
//
#include <hip/hip_runtime.h>
#include <hip/hip_bf16.h>
#include <cstdint>

typedef __attribute__((ext_vector_type(16))) _Float16 v16h;
typedef __attribute__((ext_vector_type(8)))  float    v8f;
typedef __attribute__((ext_vector_type(4)))  _Float16 v4h;
typedef __attribute__((ext_vector_type(4)))  float    v4f;

#define BATCH   128
#define NEP     64
#define DIM     128
#define MROWS   (BATCH * NEP)       // 8192 rows
#define NTILES  (MROWS / 16)        // 512 16x16 tiles per side
#define WPB     4                   // waves per block (column-split factor)
#define CT_PER_WAVE (NTILES / WPB)  // 128 column tiles per wave

// Per-wave LDS: 2 buffers x (fk slab 4KB + fq slab 4KB) = 16KB; 4 waves = 64KB.
#define SLAB_B    4096              // one 16-col x 256B slab (contiguous in memory!)
#define BUF_B     (2 * SLAB_B)      // fk + fq slab
#define WAVE_LDS_B (2 * BUF_B)      // double buffered

// ---------------------------------------------------------------------------
// Kernel 1: L2-normalize rows of fk/fq (fp32) and convert to f16.
// ---------------------------------------------------------------------------
__global__ void __launch_bounds__(256) norm_cvt_kernel(
    const float* __restrict__ fk, const float* __restrict__ fq,
    _Float16* __restrict__ fk16, _Float16* __restrict__ fq16) {
  const int lane = threadIdx.x & 31;
  const int wave = threadIdx.x >> 5;
  const int row  = blockIdx.x * 8 + wave;

  const v4f k = *(const v4f*)(fk + (size_t)row * DIM + lane * 4);
  const v4f q = *(const v4f*)(fq + (size_t)row * DIM + lane * 4);

  float sk = k[0]*k[0] + k[1]*k[1] + k[2]*k[2] + k[3]*k[3];
  float sq = q[0]*q[0] + q[1]*q[1] + q[2]*q[2] + q[3]*q[3];
  #pragma unroll
  for (int off = 16; off > 0; off >>= 1) {
    sk += __shfl_xor(sk, off, 32);
    sq += __shfl_xor(sq, off, 32);
  }
  const float ik = 1.0f / fmaxf(sqrtf(sk), 1e-12f);
  const float iq = 1.0f / fmaxf(sqrtf(sq), 1e-12f);

  v4h ok, oq;
  #pragma unroll
  for (int j = 0; j < 4; ++j) {
    ok[j] = (_Float16)(k[j] * ik);
    oq[j] = (_Float16)(q[j] * iq);
  }
  *(v4h*)(fk16 + (size_t)row * DIM + lane * 4) = ok;
  *(v4h*)(fq16 + (size_t)row * DIM + lane * 4) = oq;
}

// ---------------------------------------------------------------------------
// Kernel 2: fused contrastive loss with async global->LDS double buffering.
// One block (4 waves) per 16-row tile; waves split the 512 column tiles.
// Pipeline per wave: stage tile i+1 into LDS via global_load_async_to_lds_b128
// (ASYNCcnt), s_wait_asynccnt 16 (tile i ready, tile i+1 in flight), then
// ds_load_b128 fragments + 8x v_wmma_f32_16x16x32_f16 + exp/accumulate.
// ---------------------------------------------------------------------------
__global__ void __launch_bounds__(32 * WPB) contrast_kernel(
    const _Float16* __restrict__ fk16, const _Float16* __restrict__ fq16,
    float* __restrict__ partial) {
  __shared__ uint32_t smem4[WPB * WAVE_LDS_B / 4];   // 64 KB

  const int lane = threadIdx.x & 31;
  const int wave = threadIdx.x >> 5;
  const int t    = blockIdx.x;                // row-tile index 0..511
  const int m    = lane & 15;                 // row (A) / column (B,C) in tile
  const int h    = lane >> 4;                 // lane half

  // LDS byte address of this wave's staging area (flat addr low 32 bits ==
  // LDS byte offset per the CDNA5 flat-aperture rules).
  const uint32_t waveLdsByte =
      (uint32_t)(uintptr_t)(&smem4[0]) + (uint32_t)(wave * WAVE_LDS_B);
  const int waveLdsDw = wave * (WAVE_LDS_B / 4);

  // ---- A fragments (16x32 f16 layout), 4 K-chunks, both matrices ----
  v16h ak[4], aq[4];
  {
    const uint32_t* rowk = (const uint32_t*)(fk16 + (size_t)(t * 16 + m) * DIM);
    const uint32_t* rowq = (const uint32_t*)(fq16 + (size_t)(t * 16 + m) * DIM);
    #pragma unroll
    for (int kc = 0; kc < 4; ++kc) {
      union { v16h v; uint32_t u[8]; } tk, tq;
      const int b0 = kc * 16 + h * 8;
      const int b1 = kc * 16 + h * 8 + 4;
      #pragma unroll
      for (int j = 0; j < 4; ++j) {
        tk.u[j]     = rowk[b0 + j];  tk.u[4 + j] = rowk[b1 + j];
        tq.u[j]     = rowq[b0 + j];  tq.u[4 + j] = rowq[b1 + j];
      }
      ak[kc] = tk.v;  aq[kc] = tq.v;
    }
  }

  // Stage one column tile (16 contiguous rows = 4KB per matrix) into LDS.
  auto stage = [&](int buf, int ct) {
    const uint32_t lk = waveLdsByte + (uint32_t)(buf * BUF_B) + (uint32_t)(lane * 16);
    const uint32_t lq = lk + SLAB_B;
    const uint64_t gk = (uint64_t)(uintptr_t)(fk16 + (size_t)ct * 16 * DIM)
                      + (uint64_t)(lane * 16);
    const uint64_t gq = (uint64_t)(uintptr_t)(fq16 + (size_t)ct * 16 * DIM)
                      + (uint64_t)(lane * 16);
    #pragma unroll
    for (int j = 0; j < 8; ++j) {
      asm volatile("global_load_async_to_lds_b128 %0, %1, off"
                   :: "v"(lk + (uint32_t)(j * 512)),
                      "v"(gk + (uint64_t)(j * 512)) : "memory");
      asm volatile("global_load_async_to_lds_b128 %0, %1, off"
                   :: "v"(lq + (uint32_t)(j * 512)),
                      "v"(gq + (uint64_t)(j * 512)) : "memory");
    }
  };

  float posAcc[8], negAcc[8], diagAcc[8];
  #pragma unroll
  for (int d = 0; d < 8; ++d) { posAcc[d] = 0.f; negAcc[d] = 0.f; diagAcc[d] = 0.f; }

  const int myGrp = t >> 2;

  stage(0, wave);   // prime the pipeline

  for (int i = 0; i < CT_PER_WAVE; ++i) {
    const int ct = wave + WPB * i;
    if (i + 1 < CT_PER_WAVE) {
      stage((i + 1) & 1, wave + WPB * (i + 1));
      asm volatile("s_wait_asynccnt 16" ::: "memory");  // tile i landed in LDS
    } else {
      asm volatile("s_wait_asynccnt 0" ::: "memory");
    }

    // Fragments from LDS (ds_load_b128): column m slab row is 64 dwords.
    const int colbase = waveLdsDw + (i & 1) * (BUF_B / 4) + m * 64;
    union frag_t { v16h v; uint32_t u[8]; };
    frag_t bk[4], bq[4];
    #pragma unroll
    for (int kc = 0; kc < 4; ++kc) {
      const int b0 = colbase + kc * 16 + h * 8;
      #pragma unroll
      for (int j = 0; j < 8; ++j) {
        bk[kc].u[j] = smem4[b0 + j];
        bq[kc].u[j] = smem4[b0 + (SLAB_B / 4) + j];
      }
    }

    v8f accK = {}; v8f accQ = {};
    #pragma unroll
    for (int kc = 0; kc < 4; ++kc) {
      accK = __builtin_amdgcn_wmma_f32_16x16x32_f16(false, ak[kc], false, bk[kc].v,
                                                    (short)0, accK, false, false);
      accQ = __builtin_amdgcn_wmma_f32_16x16x32_f16(false, aq[kc], false, bq[kc].v,
                                                    (short)0, accQ, false, false);
    }

    const bool same = ((ct >> 2) == myGrp);  // uniform per wave
    #pragma unroll
    for (int d = 0; d < 8; ++d) {
      const float e = __expf(accK[d]);
      const float p = e * accQ[d];
      negAcc[d] += p;
      if (same) { posAcc[d] += e; diagAcc[d] += p; }
    }
  }

  // Reduce over the 16 columns held by each lane half (C layout: N = lane%16).
  #pragma unroll
  for (int d = 0; d < 8; ++d) {
    #pragma unroll
    for (int off = 1; off <= 8; off <<= 1) {
      posAcc[d]  += __shfl_xor(posAcc[d],  off, 32);
      negAcc[d]  += __shfl_xor(negAcc[d],  off, 32);
      diagAcc[d] += __shfl_xor(diagAcc[d], off, 32);
    }
  }

  // Cross-wave combine. The reduction area aliases wave 0's buffer, so wait
  // for ALL waves to finish consuming their buffers first.
  __syncthreads();
  float* shf = (float*)smem4;   // [0..63]=pos, [64..127]=neg, [128..191]=diag
  if ((lane & 15) == 0) {
    #pragma unroll
    for (int d = 0; d < 8; ++d) {
      const int r = h * 8 + d;
      shf[  0 + wave * 16 + r] = posAcc[d];
      shf[ 64 + wave * 16 + r] = negAcc[d];
      shf[128 + wave * 16 + r] = diagAcc[d];
    }
  }
  __syncthreads();

  if (threadIdx.x < 16) {
    const int r = threadIdx.x;
    float pos = 0.f, neg = 0.f, dia = 0.f;
    #pragma unroll
    for (int w = 0; w < WPB; ++w) {
      pos += shf[  0 + w * 16 + r];
      neg += shf[ 64 + w * 16 + r];
      dia += shf[128 + w * 16 + r];
    }
    float l = -__logf(pos / (pos + (neg - dia)));
    #pragma unroll
    for (int off = 8; off > 0; off >>= 1) l += __shfl_xor(l, off, 32);
    if (r == 0) partial[t] = l;   // sum of this tile's 16 row losses
  }
}

// ---------------------------------------------------------------------------
// Kernel 3: deterministic final reduction of 512 partials -> mean loss.
// ---------------------------------------------------------------------------
__global__ void __launch_bounds__(256) reduce_kernel(
    const float* __restrict__ partial, float* __restrict__ out) {
  __shared__ float s[256];
  const int tid = threadIdx.x;
  s[tid] = partial[tid] + partial[tid + 256];
  __syncthreads();
  #pragma unroll
  for (int off = 128; off > 0; off >>= 1) {
    if (tid < off) s[tid] += s[tid + off];
    __syncthreads();
  }
  if (tid == 0) out[0] = s[0] * (1.0f / (float)MROWS);
}

extern "C" void kernel_launch(void* const* d_in, const int* in_sizes, int n_in,
                              void* d_out, int out_size, void* d_ws, size_t ws_size,
                              hipStream_t stream) {
  (void)in_sizes; (void)n_in; (void)out_size; (void)ws_size;
  const float* fk = (const float*)d_in[0];
  const float* fq = (const float*)d_in[1];
  float* out = (float*)d_out;

  _Float16* fk16    = (_Float16*)d_ws;                         // 2 MB
  _Float16* fq16    = fk16 + (size_t)MROWS * DIM;              // 2 MB
  float*    partial = (float*)(fq16 + (size_t)MROWS * DIM);    // 2 KB

  norm_cvt_kernel<<<MROWS / 8, 256, 0, stream>>>(fk, fq, fk16, fq16);
  contrast_kernel<<<NTILES, 32 * WPB, 0, stream>>>(fk16, fq16, partial);
  reduce_kernel<<<1, 256, 0, stream>>>(partial, out);
}
